// Correlation_69234872811912
// MI455X (gfx1250) — matrix operand order repaired and checked by reading, not compiled, not run on hardware
//
#include <hip/hip_runtime.h>
#include <hip/hip_bf16.h>

typedef _Float16 half16 __attribute__((ext_vector_type(16)));
typedef _Float16 half8  __attribute__((ext_vector_type(8)));
typedef float    floatx8 __attribute__((ext_vector_type(8)));

#define C_DIM   128
#define W_DIM   128
#define H_DIM   128
#define KSZ     21
#define RAD     10
#define PITCH   136          // halves per spatial row in LDS (272B = 68 dwords, bank-friendly)
#define TGT_ROWS 160         // w' in [-16, 144)
#define GPITCH  48           // 3 tiles * 16 cols of f32 staging per wave

__device__ __forceinline__ float quantize(float x) {
    const float s = 1e10f;
    return floorf(x * s) / s;
}

__global__ void __launch_bounds__(256)
corr_wmma_kernel(const float* __restrict__ in, float* __restrict__ out) {
    extern __shared__ char smem[];
    _Float16* sSrc = (_Float16*)smem;                       // 128 * PITCH halves
    _Float16* sTgt = sSrc + C_DIM * PITCH;                  // 160 * PITCH halves
    float*    sG   = (float*)(sTgt + TGT_ROWS * PITCH);     // 8 waves * 16 * 48 f32

    const int dy = blockIdx.x;          // 0..20
    const int h  = blockIdx.y;          // 0..127
    const int b  = blockIdx.z;          // 0..3
    const int tid  = threadIdx.x;       // 0..255
    const int wave = tid >> 5;          // 0..7
    const int lane = tid & 31;

    // ---- Stage src row: global [b][0][c][h][w] -> LDS [w][c] (f16, quantized)
    {
        const size_t base = (((size_t)b * 2 + 0) * C_DIM) * (size_t)(H_DIM * W_DIM)
                          + (size_t)h * W_DIM;
        for (int i = tid; i < C_DIM * W_DIM; i += 256) {
            int c = i >> 7, w = i & 127;
            float v = in[base + (size_t)c * (H_DIM * W_DIM) + w];
            sSrc[w * PITCH + c] = (_Float16)quantize(v);
        }
    }

    // ---- Stage tgt row h' = h + dy - RAD into rows [16..144) ; zero-pad rest
    {
        const int hp = h + dy - RAD;
        if (hp >= 0 && hp < H_DIM) {
            const size_t base = (((size_t)b * 2 + 1) * C_DIM) * (size_t)(H_DIM * W_DIM)
                              + (size_t)hp * W_DIM;
            for (int i = tid; i < C_DIM * W_DIM; i += 256) {
                int c = i >> 7, w = i & 127;
                float v = in[base + (size_t)c * (H_DIM * W_DIM) + w];
                sTgt[(w + 16) * PITCH + c] = (_Float16)quantize(v);
            }
            // zero pad rows: w' in [-16,0) -> rows 0..15, w' in [128,144) -> rows 144..159
            for (int i = tid; i < 16 * PITCH; i += 256) {
                int r = i / PITCH, c = i % PITCH;
                sTgt[r * PITCH + c] = (_Float16)0.0f;
                sTgt[(144 + r) * PITCH + c] = (_Float16)0.0f;
            }
        } else {
            for (int i = tid; i < TGT_ROWS * PITCH; i += 256)
                sTgt[i] = (_Float16)0.0f;
        }
    }

    __syncthreads();

    // ---- Each wave: 16 output pixels w0..w0+15, banded Gram via 3 column tiles
    const int w0 = wave * 16;
    const int n  = lane & 15;       // M for A-frag, N for B/D-frags
    const int hs = lane >> 4;       // half-select per WMMA 16-bit layouts

    floatx8 acc0 = {}, acc1 = {}, acc2 = {};

    const _Float16* aRow  = sSrc + (size_t)(w0 + n) * PITCH;
    // tile t covers w' = w0 - 16 + 16*t + n  -> LDS row = (w' + 16)
    const _Float16* bRow0 = sTgt + (size_t)(w0 +  0 + n) * PITCH;
    const _Float16* bRow1 = sTgt + (size_t)(w0 + 16 + n) * PITCH;
    const _Float16* bRow2 = sTgt + (size_t)(w0 + 32 + n) * PITCH;

#pragma unroll
    for (int ks = 0; ks < 4; ++ks) {
        const int kc = ks * 32;
        // A fragment: halves 0..15 -> K = kc + hs*16 + i
        half8 alo = *(const half8*)(aRow + kc + hs * 16);
        half8 ahi = *(const half8*)(aRow + kc + hs * 16 + 8);
        half16 a;
#pragma unroll
        for (int i = 0; i < 8; ++i) { a[i] = alo[i]; a[i + 8] = ahi[i]; }

        // B fragments: 16 contiguous channels K = kc + hs*16 + (0..15) at fixed column
        half8 b0l = *(const half8*)(bRow0 + kc + hs * 16);
        half8 b0h = *(const half8*)(bRow0 + kc + hs * 16 + 8);
        half8 b1l = *(const half8*)(bRow1 + kc + hs * 16);
        half8 b1h = *(const half8*)(bRow1 + kc + hs * 16 + 8);
        half8 b2l = *(const half8*)(bRow2 + kc + hs * 16);
        half8 b2h = *(const half8*)(bRow2 + kc + hs * 16 + 8);
        half16 b0, b1, b2;
#pragma unroll
        for (int i = 0; i < 8; ++i) {
            b0[i] = b0l[i]; b0[i + 8] = b0h[i];
            b1[i] = b1l[i]; b1[i + 8] = b1h[i];
            b2[i] = b2l[i]; b2[i + 8] = b2h[i];
        }

        acc0 = __builtin_amdgcn_wmma_f32_16x16x32_f16(false, a, false, b0, (short)0, acc0, false, false);
        acc1 = __builtin_amdgcn_wmma_f32_16x16x32_f16(false, a, false, b1, (short)0, acc1, false, false);
        acc2 = __builtin_amdgcn_wmma_f32_16x16x32_f16(false, a, false, b2, (short)0, acc2, false, false);
    }

    // ---- Spill D tiles: lane holds N=n, VGPR r holds M = r + 8*hs
    {
        float* g = sG + wave * (16 * GPITCH);
#pragma unroll
        for (int r = 0; r < 8; ++r) {
            const int m = r + 8 * hs;
            g[m * GPITCH +  0 + n] = acc0[r];
            g[m * GPITCH + 16 + n] = acc1[r];
            g[m * GPITCH + 32 + n] = acc2[r];
        }
    }

    __syncthreads();

    // ---- Extract band: out[b, dy*21+j, h, w0+m] = G[m, m+j-10] / 128
    {
        const float* g = sG + wave * (16 * GPITCH);
        const float invC = 1.0f / 128.0f;
        for (int i = lane; i < KSZ * 16; i += 32) {
            const int j = i >> 4;           // 0..20
            const int m = i & 15;           // 0..15
            const float v = g[m * GPITCH + (m + j + 6)] * invC;   // local col = m + j - 10 + 16
            const size_t k = (size_t)dy * KSZ + j;
            out[(((size_t)b * (KSZ * KSZ) + k) * H_DIM + h) * W_DIM + (w0 + m)] = v;
        }
    }
}

extern "C" void kernel_launch(void* const* d_in, const int* in_sizes, int n_in,
                              void* d_out, int out_size, void* d_ws, size_t ws_size,
                              hipStream_t stream) {
    const float* in = (const float*)d_in[0];
    float* out = (float*)d_out;
    (void)in_sizes; (void)n_in; (void)out_size; (void)d_ws; (void)ws_size;

    dim3 grid(KSZ, H_DIM, 4);   // (dy, h, b)
    dim3 block(256);            // 8 waves
    size_t shmem = (size_t)(C_DIM + TGT_ROWS) * PITCH * sizeof(_Float16)
                 + (size_t)8 * 16 * GPITCH * sizeof(float);   // 102,912 B
    corr_wmma_kernel<<<grid, block, shmem, stream>>>(in, out);
}
